// E_DiT_Block_79955111182851
// MI455X (gfx1250) — matrix operand order repaired and checked by reading, not compiled, not run on hardware
//
#include <hip/hip_runtime.h>
#include <hip/hip_bf16.h>

// ---- problem constants ----
#define NN    32768
#define EEg   524288
#define DDim  128
#define DEdim 64
#define BBat  64
#define CUTf  5.0f

typedef __attribute__((ext_vector_type(16))) __bf16 v16bf;
typedef __attribute__((ext_vector_type(8)))  float  v8f;

// ---------------- helpers ----------------
__device__ __forceinline__ float gelu_tanh(float x) {
  const float k0 = 0.7978845608028654f;
  const float k1 = 0.044715f;
  return 0.5f * x * (1.0f + tanhf(k0 * (x + k1 * x * x * x)));
}

// swizzled A index: [rt][kt][lane][j], contiguous k order (k = 16h + j), matching B
__device__ __forceinline__ int a_sw_idx(int rt, int m, int k, int KT) {
  int kt = k >> 5, kl = k & 31;
  int j, h;
  j = kl & 15; h = kl >> 4;
  return rt * (KT << 9) + (kt << 9) + ((m + (h << 4)) << 4) + j;
}

// load W[K x Ncol] f32 row-major -> swizzled bf16 B layout [CT][KT][32][16]
__device__ __forceinline__ void load_w_sw(const float* W, __bf16* dst, int K, int Ncol, int KT) {
  int total = (Ncol >> 4) * (KT << 9);
  for (int idx = threadIdx.x; idx < total; idx += blockDim.x) {
    int j = idx & 15;
    int lane = (idx >> 4) & 31;
    int kt = (idx >> 9) % KT;
    int ct = idx / (KT << 9);
    int n = lane & 15, h = lane >> 4;
    int k = (kt << 5) + j + (h << 4);
    float v = (k < K) ? W[k * Ncol + (ct << 4) + n] : 0.0f;
    dst[idx] = (__bf16)v;
  }
}

__device__ __forceinline__ v8f wmma_kloop(const __bf16* A, const __bf16* B, int KT, v8f c) {
  int lane = threadIdx.x & 31;
  const __bf16* ap = A + lane * 16;
  const __bf16* bp = B + lane * 16;
  for (int kt = 0; kt < KT; ++kt) {
    v16bf a = *(const v16bf*)(ap + (kt << 9));
    v16bf b = *(const v16bf*)(bp + (kt << 9));
    c = __builtin_amdgcn_wmma_f32_16x16x32_bf16(false, a, false, b, (short)0, c, false, false);
  }
  return c;
}

__device__ __forceinline__ unsigned f2mono(float f) {
  unsigned u = __float_as_uint(f);
  return (u & 0x80000000u) ? ~u : (u | 0x80000000u);
}
__device__ __forceinline__ float mono2f(unsigned u) {
  return __uint_as_float((u & 0x80000000u) ? (u & 0x7fffffffu) : ~u);
}

__device__ __forceinline__ void atomicAddF32(float* p, float v) {
  asm volatile("global_atomic_add_f32 %0, %1, off"
               :: "v"((unsigned long long)(uintptr_t)p), "v"(v) : "memory");
}

// ---------------- K0: per-batch adaLN / time tables ----------------
__global__ void k_tables(const float* t,
                         const float* nn1_w, const float* nn1_b,
                         const float* nn2_w, const float* nn2_b,
                         const float* en1_w, const float* en1_b,
                         const float* en2_w, const float* en2_b,
                         const float* pu_wt, const float* pu_bt,
                         float* ss1, float* ss2, float* se1, float* se2, float* tg) {
  int b = blockIdx.x;
  __shared__ float tl[128];
  for (int i = threadIdx.x; i < 128; i += blockDim.x) tl[i] = t[b * 128 + i];
  __syncthreads();
  for (int o = threadIdx.x; o < 769; o += blockDim.x) {
    const float* w; const float* bias; float* dst; int col; int ncol;
    if (o < 256)      { w = nn1_w; bias = nn1_b; dst = ss1 + b*256; col = o;     ncol = 256; }
    else if (o < 512) { w = nn2_w; bias = nn2_b; dst = ss2 + b*256; col = o-256; ncol = 256; }
    else if (o < 640) { w = en1_w; bias = en1_b; dst = se1 + b*128; col = o-512; ncol = 128; }
    else if (o < 768) { w = en2_w; bias = en2_b; dst = se2 + b*128; col = o-640; ncol = 128; }
    else              { w = pu_wt; bias = pu_bt; dst = tg  + b;     col = 0;     ncol = 1;   }
    float acc = bias[col];
    for (int k = 0; k < 128; ++k) acc += tl[k] * w[k * ncol + col];
    dst[col] = acc;
  }
}

__global__ void k_zero(float* p, long n) {
  long i = (long)blockIdx.x * blockDim.x + threadIdx.x;
  if (i < n) p[i] = 0.0f;
}

// ---------------- K1: xn = adaLN(node_input); xs = xn@Wsrc, xd = xn@Wdst ----------------
__global__ void __launch_bounds__(256) k_node_xsxd(
    const float* node_input, const int* batch, const float* ss1,
    const float* wsrc, const float* wdst, float* xs, float* xd) {
  extern __shared__ __attribute__((aligned(32))) char smem[];
  __bf16* Wsw = (__bf16*)smem;            // 2 * 8ct * 4kt * 512
  __bf16* Asw = Wsw + 2 * 8 * 4 * 512;    // 4rt * 4kt * 512
  load_w_sw(wsrc, Wsw,             128, 128, 4);
  load_w_sw(wdst, Wsw + 8*4*512,   128, 128, 4);
  int rowBase = blockIdx.x * 64;
  int tid = threadIdx.x;
  int r = tid >> 2, q = tid & 3;
  int g = rowBase + r;
  int rt = r >> 4, m = r & 15;
  float vals[32]; float s = 0.f, s2 = 0.f;
  #pragma unroll
  for (int i = 0; i < 32; ++i) {
    float v = node_input[(size_t)g * DDim + q*32 + i];
    vals[i] = v; s += v; s2 += v * v;
  }
  s  += __shfl_xor(s, 1);  s  += __shfl_xor(s, 2);
  s2 += __shfl_xor(s2, 1); s2 += __shfl_xor(s2, 2);
  float mu = s / 128.f, var = s2 / 128.f - mu * mu;
  float rstd = rsqrtf(var + 1e-6f);
  const float* ssr = ss1 + batch[g] * 256;
  #pragma unroll
  for (int i = 0; i < 32; ++i) {
    int c = q*32 + i;
    float xn = (vals[i] - mu) * rstd;
    Asw[a_sw_idx(rt, m, c, 4)] = (__bf16)(xn * (1.f + ssr[c]) + ssr[128 + c]);
  }
  __syncthreads();
  int wave = tid >> 5, lane = tid & 31, n16 = lane & 15, hh = lane >> 4;
  for (int i = 0; i < 8; ++i) {
    int idx = wave * 8 + i;          // 0..63
    int mat = idx >> 5;
    int rt2 = (idx >> 3) & 3;
    int ct  = idx & 7;
    v8f c = {};
    c = wmma_kloop(Asw + rt2*4*512, Wsw + mat*(8*4*512) + ct*4*512, 4, c);
    float* out = mat ? xd : xs;
    #pragma unroll
    for (int rr = 0; rr < 8; ++rr)
      out[(size_t)(rowBase + rt2*16 + rr + hh*8) * DDim + ct*16 + n16] = c[rr];
  }
}

// ---------------- K2: edge message + filter + alpha (+ amax) ----------------
__global__ void __launch_bounds__(256) k_edge_msg(
    const int* edge_src, const int* edge_dst, const float* edge_input,
    const float* pos, const float* xs, const float* xd,
    const float* ga_we, const float* ga_wsh,
    const float* ga_fc1, const float* ga_fb1,
    const float* ga_fc2, const float* ga_fb2,
    const float* ga_a,
    __bf16* msg_out, float* alpha_out, unsigned* amax) {
  extern __shared__ __attribute__((aligned(32))) char smem[];
  __bf16* Wcb = (__bf16*)smem;     // 8ct*3kt*512   (ein|sh combined, Kpad=96)
  __bf16* Wf1 = Wcb + 12288;       // 4ct*1kt*512
  __bf16* Wf2 = Wf1 + 2048;        // 8ct*2kt*512
  __bf16* Aes = Wf2 + 8192;        // 4rt*3kt*512
  __bf16* Arb = Aes + 6144;        // 4rt*1kt*512
  __bf16* Ahd = Arb + 2048;        // 4rt*2kt*512
  float*  msgf = (float*)(Ahd + 4096);   // 64*128
  int* sidx = (int*)(msgf + 8192);
  int* didx = sidx + 64;

  int tid = threadIdx.x;
  for (int idx = tid; idx < 8*3*512; idx += 256) {   // combined weight
    int j = idx & 15, lane = (idx >> 4) & 31, kt = (idx >> 9) % 3, ct = idx / (3*512);
    int n = lane & 15, h = lane >> 4;
    int k = kt*32 + j + 16*h;
    float v = 0.f;
    if (k < 64)      v = ga_we[k * DDim + ct*16 + n];
    else if (k < 73) v = ga_wsh[(k - 64) * DDim + ct*16 + n];
    Wcb[idx] = (__bf16)v;
  }
  load_w_sw(ga_fc1, Wf1, 32, 64, 1);
  load_w_sw(ga_fc2, Wf2, 64, 128, 2);

  int eb = blockIdx.x * 64;
  if (tid < 64) {
    int e = eb + tid;
    int s = edge_src[e], d = edge_dst[e];
    sidx[tid] = s; didx[tid] = d;
    float rx = pos[s*3+0]-pos[d*3+0], ry = pos[s*3+1]-pos[d*3+1], rz = pos[s*3+2]-pos[d*3+2];
    float dist = fmaxf(sqrtf(rx*rx + ry*ry + rz*rz), 1e-8f);
    float ux = rx/dist, uy = ry/dist, uz = rz/dist;
    const float s15 = 3.8729833462074170f, s5 = 2.2360679774997896f, s3 = 1.7320508075688772f;
    float sh[9];
    sh[0]=1.f; sh[1]=s3*ux; sh[2]=s3*uy; sh[3]=s3*uz;
    sh[4]=s15*ux*uy; sh[5]=s15*uy*uz; sh[6]=0.5f*s5*(3.f*uz*uz-1.f);
    sh[7]=s15*ux*uz; sh[8]=0.5f*s15*(ux*ux-uy*uy);
    int rt = tid >> 4, m = tid & 15;
    #pragma unroll
    for (int i = 0; i < 9; ++i) Aes[a_sw_idx(rt, m, 64+i, 3)] = (__bf16)sh[i];
    for (int k = 73; k < 96; ++k) Aes[a_sw_idx(rt, m, k, 3)] = (__bf16)0.f;
    const float width = CUTf / 32.f;
    #pragma unroll
    for (int i = 0; i < 32; ++i) {
      float cen = CUTf * (float)i / 31.f;
      float z = (dist - cen) / width;
      Arb[a_sw_idx(rt, m, i, 1)] = (__bf16)__expf(-z*z);
    }
  }
  {
    int el = tid >> 2, q = tid & 3;
    int e = eb + el;
    int rt = el >> 4, m = el & 15;
    #pragma unroll
    for (int i = 0; i < 16; ++i) {
      int k = q*16 + i;
      Aes[a_sw_idx(rt, m, k, 3)] = (__bf16)edge_input[(size_t)e * DEdim + k];
    }
  }
  __syncthreads();
  int wave = tid >> 5, lane = tid & 31, n16 = lane & 15, hh = lane >> 4;
  for (int i = 0; i < 4; ++i) {                       // msg pre
    int idx = wave*4 + i;
    int rt = idx >> 3, ct = idx & 7;
    v8f c = {};
    c = wmma_kloop(Aes + rt*3*512, Wcb + ct*3*512, 3, c);
    #pragma unroll
    for (int rr = 0; rr < 8; ++rr) {
      int row = rt*16 + rr + hh*8, col = ct*16 + n16;
      msgf[row*128 + col] = c[rr] + xs[(size_t)sidx[row]*DDim + col] + xd[(size_t)didx[row]*DDim + col];
    }
  }
  for (int i = 0; i < 2; ++i) {                       // hid = gelu(rbf@fc1+b)
    int idx = wave*2 + i;
    int rt = idx >> 2, ct = idx & 3;
    v8f c = {};
    c = wmma_kloop(Arb + rt*512, Wf1 + ct*512, 1, c);
    #pragma unroll
    for (int rr = 0; rr < 8; ++rr) {
      int mrow = rr + hh*8, col = ct*16 + n16;
      Ahd[a_sw_idx(rt, mrow, col, 2)] = (__bf16)gelu_tanh(c[rr] + ga_fb1[col]);
    }
  }
  __syncthreads();
  for (int i = 0; i < 4; ++i) {                       // msg *= hid@fc2+b
    int idx = wave*4 + i;
    int rt = idx >> 3, ct = idx & 7;
    v8f c = {};
    c = wmma_kloop(Ahd + rt*2*512, Wf2 + ct*2*512, 2, c);
    #pragma unroll
    for (int rr = 0; rr < 8; ++rr) {
      int row = rt*16 + rr + hh*8, col = ct*16 + n16;
      float v = msgf[row*128 + col] * (c[rr] + ga_fb2[col]);
      msgf[row*128 + col] = v;
      msg_out[(size_t)(eb + row) * DDim + col] = (__bf16)v;
    }
  }
  __syncthreads();
  {
    int el = tid >> 2, q = tid & 3;
    int e = eb + el;
    int d = didx[el];
    #pragma unroll
    for (int hi = 0; hi < 2; ++hi) {
      int h = q*2 + hi;
      float acc = 0.f;
      #pragma unroll
      for (int dd = 0; dd < 16; ++dd) {
        float x = msgf[el*128 + h*16 + dd];
        acc += (x > 0.f ? x : 0.2f * x) * ga_a[h*16 + dd];
      }
      alpha_out[(size_t)e*8 + h] = acc;
      atomicMax(&amax[d*8 + h], f2mono(acc));
    }
  }
}

// ---------------- K3/K4: softmax denom, aggregation ----------------
__global__ void k_softmax_den(float* alpha_ex, const int* edge_dst, const unsigned* amax, float* den) {
  long i = (long)blockIdx.x * blockDim.x + threadIdx.x;
  if (i >= (long)EEg * 8) return;
  int e = (int)(i >> 3), h = (int)(i & 7);
  int d = edge_dst[e];
  float v = __expf(alpha_ex[i] - mono2f(amax[d*8 + h]));
  alpha_ex[i] = v;
  atomicAddF32(&den[d*8 + h], v);
}

__global__ void k_aggregate(const __bf16* msg, const float* ex, const float* den,
                            const int* edge_dst, float* agg) {
  long i = (long)blockIdx.x * blockDim.x + threadIdx.x;
  if (i >= (long)EEg * 128) return;
  int e = (int)(i >> 7), c = (int)(i & 127), h = c >> 4;
  int d = edge_dst[e];
  float attn = ex[(size_t)e*8 + h] / (den[d*8 + h] + 1e-9f);
  atomicAddF32(&agg[(size_t)d*128 + c], (float)msg[i] * attn);
}

// ---------------- K5: ga_out + residual, adaLN2 + node_attr proj ----------------
__global__ void __launch_bounds__(256) k_node2a(
    const float* agg, const float* node_input, const float* node_attr,
    const int* batch, const float* ss2,
    const float* ga_wo, const float* ga_bo, const float* nf_wa,
    const float* gate_ga, float* out_node, __bf16* hin) {
  extern __shared__ __attribute__((aligned(32))) char smem[];
  __bf16* Wo  = (__bf16*)smem;     // 8*4*512
  __bf16* Wa  = Wo + 16384;        // 8*1*512
  __bf16* Aag = Wa + 4096;         // 4*4*512
  __bf16* Aat = Aag + 8192;        // 4*1*512
  float* nof = (float*)(Aat + 2048);  // 64*128
  float* gf  = nof + 8192;            // 64*128
  load_w_sw(ga_wo, Wo, 128, 128, 4);
  load_w_sw(nf_wa, Wa, 16, 128, 1);
  int rowBase = blockIdx.x * 64;
  int tid = threadIdx.x;
  int r = tid >> 2, q = tid & 3;
  int g = rowBase + r;
  int rt = r >> 4, m = r & 15;
  #pragma unroll
  for (int i = 0; i < 32; ++i) {
    int c = q*32 + i;
    Aag[a_sw_idx(rt, m, c, 4)] = (__bf16)agg[(size_t)g*128 + c];
  }
  #pragma unroll
  for (int i = 0; i < 4; ++i) {
    int c = q*4 + i;
    Aat[a_sw_idx(rt, m, c, 1)]      = (__bf16)node_attr[(size_t)g*16 + c];
    Aat[a_sw_idx(rt, m, c + 16, 1)] = (__bf16)0.f;
  }
  __syncthreads();
  int wave = tid >> 5, lane = tid & 31, n16 = lane & 15, hh = lane >> 4;
  float gga = gate_ga[0];
  for (int i = 0; i < 4; ++i) {
    int idx = wave*4 + i;
    int rt2 = idx >> 3, ct = idx & 7;
    v8f c = {};
    c = wmma_kloop(Aag + rt2*4*512, Wo + ct*4*512, 4, c);
    #pragma unroll
    for (int rr = 0; rr < 8; ++rr) {
      int row = rt2*16 + rr + hh*8, col = ct*16 + n16;
      float no = node_input[(size_t)(rowBase+row)*128 + col] + gga * (c[rr] + ga_bo[col]);
      out_node[(size_t)(rowBase+row)*128 + col] = no;
      nof[row*128 + col] = no;
    }
  }
  for (int i = 0; i < 4; ++i) {
    int idx = wave*4 + i;
    int rt2 = idx >> 3, ct = idx & 7;
    v8f c = {};
    c = wmma_kloop(Aat + rt2*512, Wa + ct*512, 1, c);
    #pragma unroll
    for (int rr = 0; rr < 8; ++rr) {
      int row = rt2*16 + rr + hh*8, col = ct*16 + n16;
      gf[row*128 + col] = c[rr];
    }
  }
  __syncthreads();
  float s = 0.f, s2 = 0.f; float vals[32];
  #pragma unroll
  for (int i = 0; i < 32; ++i) {
    float v = nof[r*128 + q*32 + i];
    vals[i] = v; s += v; s2 += v*v;
  }
  s  += __shfl_xor(s, 1);  s  += __shfl_xor(s, 2);
  s2 += __shfl_xor(s2, 1); s2 += __shfl_xor(s2, 2);
  float mu = s / 128.f, rstd = rsqrtf(s2 / 128.f - mu*mu + 1e-6f);
  const float* ssr = ss2 + batch[g] * 256;
  #pragma unroll
  for (int i = 0; i < 32; ++i) {
    int c = q*32 + i;
    float xn = (vals[i] - mu) * rstd;
    hin[(size_t)g*128 + c] = (__bf16)(xn * (1.f + ssr[c]) + ssr[128 + c] + gf[r*128 + c]);
  }
}

// ---------------- K6/K7: node FFN ----------------
__global__ void __launch_bounds__(256) k_ffn1(
    const __bf16* hin, const float* nf_w1, const float* nf_b1, __bf16* hbuf) {
  extern __shared__ __attribute__((aligned(32))) char smem[];
  __bf16* W1 = (__bf16*)smem;   // 32*4*512
  __bf16* A  = W1 + 65536;      // 4*4*512
  load_w_sw(nf_w1, W1, 128, 512, 4);
  int rowBase = blockIdx.x * 64;
  int tid = threadIdx.x;
  int r = tid >> 2, q = tid & 3, g = rowBase + r;
  int rt = r >> 4, m = r & 15;
  #pragma unroll
  for (int i = 0; i < 32; ++i) {
    int c = q*32 + i;
    A[a_sw_idx(rt, m, c, 4)] = hin[(size_t)g*128 + c];
  }
  __syncthreads();
  int wave = tid >> 5, lane = tid & 31, n16 = lane & 15, hh = lane >> 4;
  for (int i = 0; i < 16; ++i) {
    int idx = wave*16 + i;   // 0..127
    int rt2 = idx >> 5, ct = idx & 31;
    v8f c = {};
    c = wmma_kloop(A + rt2*4*512, W1 + ct*4*512, 4, c);
    #pragma unroll
    for (int rr = 0; rr < 8; ++rr) {
      int row = rt2*16 + rr + hh*8, col = ct*16 + n16;
      hbuf[(size_t)(rowBase+row)*512 + col] = (__bf16)gelu_tanh(c[rr] + nf_b1[col]);
    }
  }
}

__global__ void __launch_bounds__(256) k_ffn2(
    const __bf16* hbuf, const float* nf_w2, const float* nf_b2,
    const float* gate_nffn, float* out_node) {
  extern __shared__ __attribute__((aligned(32))) char smem[];
  __bf16* W2 = (__bf16*)smem;  // 8*16*512
  __bf16* A  = W2 + 65536;     // 4*16*512
  load_w_sw(nf_w2, W2, 512, 128, 16);
  int rowBase = blockIdx.x * 64;
  int tid = threadIdx.x;
  int r = tid >> 2, q = tid & 3, g = rowBase + r;
  int rt = r >> 4, m = r & 15;
  for (int i = 0; i < 128; ++i) {
    int c = q*128 + i;
    A[a_sw_idx(rt, m, c, 16)] = hbuf[(size_t)g*512 + c];
  }
  __syncthreads();
  int wave = tid >> 5, lane = tid & 31, n16 = lane & 15, hh = lane >> 4;
  float gn = gate_nffn[0];
  for (int i = 0; i < 4; ++i) {
    int idx = wave*4 + i;
    int rt2 = idx >> 3, ct = idx & 7;
    v8f c = {};
    c = wmma_kloop(A + rt2*16*512, W2 + ct*16*512, 16, c);
    #pragma unroll
    for (int rr = 0; rr < 8; ++rr) {
      int row = rt2*16 + rr + hh*8, col = ct*16 + n16;
      size_t o = (size_t)(rowBase+row)*128 + col;
      out_node[o] = out_node[o] + gn * (c[rr] + nf_b2[col]);
    }
  }
}

// ---------------- K8: edge update (feat 361 -> 128 -> 64) ----------------
__global__ void __launch_bounds__(256) k_edge_upd(
    const int* edge_src, const int* edge_dst, const float* edge_input,
    const float* pos, const float* out_node, const int* batch, const float* se1,
    const float* eu_w1, const float* eu_b1, const float* eu_w2, const float* eu_b2,
    const float* gate_eu, float* out_edge) {
  extern __shared__ __attribute__((aligned(32))) char smem[];
  __bf16* W1 = (__bf16*)smem;   // 8*12*512
  __bf16* W2 = W1 + 49152;      // 4*4*512
  __bf16* Af = W2 + 8192;       // 4*12*512
  __bf16* Ah = Af + 24576;      // 4*4*512
  int* sidx = (int*)(Ah + 8192);
  int* didx = sidx + 64;
  load_w_sw(eu_w1, W1, 361, 128, 12);
  load_w_sw(eu_w2, W2, 128, 64, 4);
  int eb = blockIdx.x * 64;
  int tid = threadIdx.x;
  if (tid < 64) {
    int e = eb + tid;
    int s = edge_src[e], d = edge_dst[e];
    sidx[tid] = s; didx[tid] = d;
    float rx = pos[s*3+0]-pos[d*3+0], ry = pos[s*3+1]-pos[d*3+1], rz = pos[s*3+2]-pos[d*3+2];
    float dist = fmaxf(sqrtf(rx*rx + ry*ry + rz*rz), 1e-8f);
    float ux = rx/dist, uy = ry/dist, uz = rz/dist;
    const float s15 = 3.8729833462074170f, s5 = 2.2360679774997896f, s3 = 1.7320508075688772f;
    float sh[9];
    sh[0]=1.f; sh[1]=s3*ux; sh[2]=s3*uy; sh[3]=s3*uz;
    sh[4]=s15*ux*uy; sh[5]=s15*uy*uz; sh[6]=0.5f*s5*(3.f*uz*uz-1.f);
    sh[7]=s15*ux*uz; sh[8]=0.5f*s15*(ux*ux-uy*uy);
    int rt = tid >> 4, m = tid & 15;
    const float width = CUTf / 32.f;
    #pragma unroll
    for (int i = 0; i < 32; ++i) {
      float cen = CUTf * (float)i / 31.f;
      float z = (dist - cen) / width;
      Af[a_sw_idx(rt, m, 320 + i, 12)] = (__bf16)__expf(-z*z);
    }
    #pragma unroll
    for (int i = 0; i < 9; ++i) Af[a_sw_idx(rt, m, 352 + i, 12)] = (__bf16)sh[i];
    for (int k = 361; k < 384; ++k) Af[a_sw_idx(rt, m, k, 12)] = (__bf16)0.f;
  }
  __syncthreads();
  int r = tid >> 2, q = tid & 3;
  int e = eb + r;
  int rt = r >> 4, m = r & 15;
  int s = sidx[r], d = didx[r];
  #pragma unroll
  for (int i = 0; i < 32; ++i) {
    int c = q*32 + i;
    Af[a_sw_idx(rt, m, c, 12)]       = (__bf16)out_node[(size_t)s*128 + c];
    Af[a_sw_idx(rt, m, 128 + c, 12)] = (__bf16)out_node[(size_t)d*128 + c];
  }
  float sum = 0.f, sum2 = 0.f; float ev[16];
  #pragma unroll
  for (int i = 0; i < 16; ++i) {
    float v = edge_input[(size_t)e*64 + q*16 + i];
    ev[i] = v; sum += v; sum2 += v*v;
  }
  sum  += __shfl_xor(sum, 1);  sum  += __shfl_xor(sum, 2);
  sum2 += __shfl_xor(sum2, 1); sum2 += __shfl_xor(sum2, 2);
  float mu = sum / 64.f, rstd = rsqrtf(sum2 / 64.f - mu*mu + 1e-6f);
  const float* ssr = se1 + batch[s] * 128;
  #pragma unroll
  for (int i = 0; i < 16; ++i) {
    int c = q*16 + i;
    float xn = (ev[i] - mu) * rstd;
    Af[a_sw_idx(rt, m, 256 + c, 12)] = (__bf16)(xn * (1.f + ssr[c]) + ssr[64 + c]);
  }
  __syncthreads();
  int wave = tid >> 5, lane = tid & 31, n16 = lane & 15, hh = lane >> 4;
  for (int i = 0; i < 4; ++i) {
    int idx = wave*4 + i;
    int rt2 = idx >> 3, ct = idx & 7;
    v8f c = {};
    c = wmma_kloop(Af + rt2*12*512, W1 + ct*12*512, 12, c);
    #pragma unroll
    for (int rr = 0; rr < 8; ++rr) {
      int col = ct*16 + n16;
      Ah[a_sw_idx(rt2, rr + hh*8, col, 4)] = (__bf16)gelu_tanh(c[rr] + eu_b1[col]);
    }
  }
  __syncthreads();
  float geu = gate_eu[0];
  for (int i = 0; i < 2; ++i) {
    int idx = wave*2 + i;
    int rt2 = idx >> 2, ct = idx & 3;
    v8f c = {};
    c = wmma_kloop(Ah + rt2*4*512, W2 + ct*4*512, 4, c);
    #pragma unroll
    for (int rr = 0; rr < 8; ++rr) {
      int row = rt2*16 + rr + hh*8, col = ct*16 + n16;
      size_t o = (size_t)(eb + row)*64 + col;
      out_edge[o] = edge_input[o] + geu * (c[rr] + eu_b2[col]);
    }
  }
}

// ---------------- K9: edge FFN ----------------
__global__ void __launch_bounds__(256) k_edge_ffn(
    const int* edge_src, const float* edge_attr, const int* batch, const float* se2,
    const float* ef_wa, const float* ef_w1, const float* ef_b1,
    const float* ef_w2, const float* ef_b2, const float* gate_effn,
    float* out_edge) {
  extern __shared__ __attribute__((aligned(32))) char smem[];
  __bf16* Wa  = (__bf16*)smem;   // 4*1*512
  __bf16* W1  = Wa + 2048;       // 16*2*512
  __bf16* W2  = W1 + 16384;      // 4*8*512
  __bf16* Aat = W2 + 16384;      // 4*1*512
  __bf16* Ax  = Aat + 2048;      // 4*2*512
  __bf16* Ah  = Ax + 4096;       // 4*8*512
  float* midf = (float*)(Ah + 16384);  // 64*64
  float* gef  = midf + 4096;           // 64*64
  float* mus  = gef + 4096;            // 64
  float* rstds = mus + 64;             // 64
  int* ebat = (int*)(rstds + 64);      // 64
  load_w_sw(ef_wa, Wa, 16, 64, 1);
  load_w_sw(ef_w1, W1, 64, 256, 2);
  load_w_sw(ef_w2, W2, 256, 64, 8);
  int eb = blockIdx.x * 64;
  int tid = threadIdx.x;
  int r = tid >> 2, q = tid & 3;
  int e = eb + r;
  int rt = r >> 4, m = r & 15;
  if (tid < 64) ebat[tid] = batch[edge_src[eb + tid]];
  float sum = 0.f, sum2 = 0.f;
  #pragma unroll
  for (int i = 0; i < 16; ++i) {
    int c = q*16 + i;
    float v = out_edge[(size_t)e*64 + c];
    midf[r*64 + c] = v; sum += v; sum2 += v*v;
  }
  sum  += __shfl_xor(sum, 1);  sum  += __shfl_xor(sum, 2);
  sum2 += __shfl_xor(sum2, 1); sum2 += __shfl_xor(sum2, 2);
  if (q == 0) { float mu = sum / 64.f; mus[r] = mu; rstds[r] = rsqrtf(sum2 / 64.f - mu*mu + 1e-6f); }
  #pragma unroll
  for (int i = 0; i < 4; ++i) {
    int c = q*4 + i;
    Aat[a_sw_idx(rt, m, c, 1)]      = (__bf16)edge_attr[(size_t)e*16 + c];
    Aat[a_sw_idx(rt, m, c + 16, 1)] = (__bf16)0.f;
  }
  __syncthreads();
  int wave = tid >> 5, lane = tid & 31, n16 = lane & 15, hh = lane >> 4;
  for (int i = 0; i < 2; ++i) {                      // ge = attr@Wa
    int idx = wave*2 + i;
    int rt2 = idx >> 2, ct = idx & 3;
    v8f c = {};
    c = wmma_kloop(Aat + rt2*512, Wa + ct*512, 1, c);
    #pragma unroll
    for (int rr = 0; rr < 8; ++rr) {
      int row = rt2*16 + rr + hh*8, col = ct*16 + n16;
      gef[row*64 + col] = c[rr];
    }
  }
  __syncthreads();
  {
    float mu = mus[r], rstd = rstds[r];
    const float* ssr = se2 + ebat[r] * 128;
    #pragma unroll
    for (int i = 0; i < 16; ++i) {
      int c = q*16 + i;
      float xn = (midf[r*64 + c] - mu) * rstd;
      Ax[a_sw_idx(rt, m, c, 2)] = (__bf16)(xn * (1.f + ssr[c]) + ssr[64 + c] + gef[r*64 + c]);
    }
  }
  __syncthreads();
  for (int i = 0; i < 8; ++i) {                      // h = gelu(x@W1+b)
    int idx = wave*8 + i;
    int rt2 = idx >> 4, ct = idx & 15;
    v8f c = {};
    c = wmma_kloop(Ax + rt2*2*512, W1 + ct*2*512, 2, c);
    #pragma unroll
    for (int rr = 0; rr < 8; ++rr) {
      int col = ct*16 + n16;
      Ah[a_sw_idx(rt2, rr + hh*8, col, 8)] = (__bf16)gelu_tanh(c[rr] + ef_b1[col]);
    }
  }
  __syncthreads();
  float ge2 = gate_effn[0];
  for (int i = 0; i < 2; ++i) {                      // effn + residual
    int idx = wave*2 + i;
    int rt2 = idx >> 2, ct = idx & 3;
    v8f c = {};
    c = wmma_kloop(Ah + rt2*8*512, W2 + ct*8*512, 8, c);
    #pragma unroll
    for (int rr = 0; rr < 8; ++rr) {
      int row = rt2*16 + rr + hh*8, col = ct*16 + n16;
      out_edge[(size_t)(eb + row)*64 + col] = midf[row*64 + col] + ge2 * (c[rr] + ef_b2[col]);
    }
  }
}

// ---------------- K10/K11: position update ----------------
__global__ void __launch_bounds__(256) k_pos_edge(
    const int* edge_src, const int* edge_dst, const float* out_edge, const float* pos,
    const float* pu_w1, const float* pu_b1, const float* pu_w2, const float* pu_b2,
    float* delta) {
  __shared__ float w1[65*32];
  __shared__ float w2[32];
  __shared__ float b1[32];
  for (int i = threadIdx.x; i < 65*32; i += blockDim.x) w1[i] = pu_w1[i];
  if (threadIdx.x < 32) { w2[threadIdx.x] = pu_w2[threadIdx.x]; b1[threadIdx.x] = pu_b1[threadIdx.x]; }
  __syncthreads();
  int e = blockIdx.x * blockDim.x + threadIdx.x;
  if (e >= EEg) return;
  int s = edge_src[e], d = edge_dst[e];
  float rx = pos[s*3+0]-pos[d*3+0], ry = pos[s*3+1]-pos[d*3+1], rz = pos[s*3+2]-pos[d*3+2];
  float dist = fmaxf(sqrtf(rx*rx + ry*ry + rz*rz), 1e-8f);
  float ux = rx/dist, uy = ry/dist, uz = rz/dist;
  float eo[64];
  #pragma unroll
  for (int k = 0; k < 64; ++k) eo[k] = out_edge[(size_t)e*64 + k];
  float sacc = pu_b2[0];
  for (int hh = 0; hh < 32; ++hh) {
    float acc = b1[hh];
    #pragma unroll 8
    for (int k = 0; k < 64; ++k) acc += eo[k] * w1[k*32 + hh];
    acc += dist * w1[64*32 + hh];
    sacc += gelu_tanh(acc) * w2[hh];
  }
  float sv = tanhf(sacc);
  atomicAddF32(&delta[d*3 + 0], sv * ux);
  atomicAddF32(&delta[d*3 + 1], sv * uy);
  atomicAddF32(&delta[d*3 + 2], sv * uz);
}

__global__ void k_pos_final(const float* pos, const float* delta, const float* tg,
                            const int* batch, const int* tmask, float* out_pos) {
  int n = blockIdx.x * blockDim.x + threadIdx.x;
  if (n >= NN) return;
  float tv = tg[batch[n]];
  int msk = tmask[n];
  #pragma unroll
  for (int i = 0; i < 3; ++i) {
    float dv = delta[n*3 + i] * tv;
    out_pos[n*3 + i] = pos[n*3 + i] + (msk ? dv : 0.f);
  }
}

// ---------------- host launcher ----------------
extern "C" void kernel_launch(void* const* d_in, const int* in_sizes, int n_in,
                              void* d_out, int out_size, void* d_ws, size_t ws_size,
                              hipStream_t stream) {
  (void)in_sizes; (void)n_in; (void)out_size; (void)ws_size;
  const float* node_input = (const float*)d_in[0];
  const float* node_attr  = (const float*)d_in[1];
  const int*   edge_src   = (const int*)d_in[2];
  const int*   edge_dst   = (const int*)d_in[3];
  const float* edge_input = (const float*)d_in[4];
  const float* edge_attr  = (const float*)d_in[5];
  const float* pos        = (const float*)d_in[6];
  const float* t          = (const float*)d_in[7];
  const int*   batch      = (const int*)d_in[8];
  const int*   tmask      = (const int*)d_in[9];
  const float* nn1_w = (const float*)d_in[10]; const float* nn1_b = (const float*)d_in[11];
  const float* nn2_w = (const float*)d_in[12]; const float* nn2_b = (const float*)d_in[13];
  const float* en1_w = (const float*)d_in[14]; const float* en1_b = (const float*)d_in[15];
  const float* en2_w = (const float*)d_in[16]; const float* en2_b = (const float*)d_in[17];
  const float* ga_wsrc = (const float*)d_in[18];
  const float* ga_wdst = (const float*)d_in[19];
  const float* ga_we   = (const float*)d_in[20];
  const float* ga_wsh  = (const float*)d_in[21];
  const float* ga_fc1  = (const float*)d_in[22]; const float* ga_fb1 = (const float*)d_in[23];
  const float* ga_fc2  = (const float*)d_in[24]; const float* ga_fb2 = (const float*)d_in[25];
  const float* ga_a    = (const float*)d_in[26];
  const float* ga_wo   = (const float*)d_in[27]; const float* ga_bo = (const float*)d_in[28];
  const float* nf_wa = (const float*)d_in[29];
  const float* nf_w1 = (const float*)d_in[30]; const float* nf_b1 = (const float*)d_in[31];
  const float* nf_w2 = (const float*)d_in[32]; const float* nf_b2 = (const float*)d_in[33];
  const float* eu_w1 = (const float*)d_in[34]; const float* eu_b1 = (const float*)d_in[35];
  const float* eu_w2 = (const float*)d_in[36]; const float* eu_b2 = (const float*)d_in[37];
  const float* ef_wa = (const float*)d_in[38];
  const float* ef_w1 = (const float*)d_in[39]; const float* ef_b1 = (const float*)d_in[40];
  const float* ef_w2 = (const float*)d_in[41]; const float* ef_b2 = (const float*)d_in[42];
  const float* pu_w1 = (const float*)d_in[43]; const float* pu_b1 = (const float*)d_in[44];
  const float* pu_w2 = (const float*)d_in[45]; const float* pu_b2 = (const float*)d_in[46];
  const float* pu_wt = (const float*)d_in[47]; const float* pu_bt = (const float*)d_in[48];
  const float* gate_ga   = (const float*)d_in[49];
  const float* gate_nffn = (const float*)d_in[50];
  const float* gate_eu   = (const float*)d_in[51];
  const float* gate_effn = (const float*)d_in[52];

  // output regions
  float* out_node = (float*)d_out;
  float* out_edge = out_node + (size_t)NN * 128;
  float* out_pos  = out_edge + (size_t)EEg * 64;

  // workspace carve
  float* ss1 = (float*)d_ws;
  float* ss2 = ss1 + 16384;
  float* se1 = ss2 + 16384;
  float* se2 = se1 + 8192;
  float* tg  = se2 + 8192;
  unsigned* amax = (unsigned*)(tg + 64);
  float* den   = (float*)(amax + (size_t)NN * 8);
  float* agg   = den + (size_t)NN * 8;
  float* delta = agg + (size_t)NN * 128;
  float* xs    = delta + (size_t)NN * 3;
  float* xd    = xs + (size_t)NN * 128;
  float* exb   = xd + (size_t)NN * 128;       // alpha then ex (in place)
  __bf16* hin  = (__bf16*)(exb + (size_t)EEg * 8);
  __bf16* hbuf = hin + (size_t)NN * 128;
  __bf16* msg  = hbuf + (size_t)NN * 512;

  // K0: tables
  k_tables<<<BBat, 256, 0, stream>>>(t, nn1_w, nn1_b, nn2_w, nn2_b, en1_w, en1_b,
                                     en2_w, en2_b, pu_wt, pu_bt, ss1, ss2, se1, se2, tg);
  // zero amax/den/agg/delta (contiguous)
  long zn = (long)NN*8 + (long)NN*8 + (long)NN*128 + (long)NN*3;
  k_zero<<<(int)((zn + 255) / 256), 256, 0, stream>>>((float*)amax, zn);

  // K1
  k_node_xsxd<<<NN/64, 256, 81920, stream>>>(node_input, batch, ss1, ga_wsrc, ga_wdst, xs, xd);
  // K2
  k_edge_msg<<<EEg/64, 256, 102912, stream>>>(edge_src, edge_dst, edge_input, pos, xs, xd,
                                              ga_we, ga_wsh, ga_fc1, ga_fb1, ga_fc2, ga_fb2,
                                              ga_a, msg, exb, amax);
  // K3
  k_softmax_den<<<(int)(((long)EEg*8 + 255) / 256), 256, 0, stream>>>(exb, edge_dst, amax, den);
  // K4
  k_aggregate<<<(int)(((long)EEg*128 + 255) / 256), 256, 0, stream>>>(msg, exb, den, edge_dst, agg);
  // K5
  k_node2a<<<NN/64, 256, 126976, stream>>>(agg, node_input, node_attr, batch, ss2,
                                           ga_wo, ga_bo, nf_wa, gate_ga, out_node, hin);
  // K6/K7
  k_ffn1<<<NN/64, 256, 147456, stream>>>(hin, nf_w1, nf_b1, hbuf);
  k_ffn2<<<NN/64, 256, 196608, stream>>>(hbuf, nf_w2, nf_b2, gate_nffn, out_node);
  // K8
  k_edge_upd<<<EEg/64, 256, 180736, stream>>>(edge_src, edge_dst, edge_input, pos, out_node,
                                              batch, se1, eu_w1, eu_b1, eu_w2, eu_b2,
                                              gate_eu, out_edge);
  // K9
  k_edge_ffn<<<EEg/64, 256, 148736, stream>>>(edge_src, edge_attr, batch, se2, ef_wa,
                                              ef_w1, ef_b1, ef_w2, ef_b2, gate_effn, out_edge);
  // K10/K11
  k_pos_edge<<<EEg/256, 256, 0, stream>>>(edge_src, edge_dst, out_edge, pos,
                                          pu_w1, pu_b1, pu_w2, pu_b2, delta);
  k_pos_final<<<(NN + 255)/256, 256, 0, stream>>>(pos, delta, tg, batch, tmask, out_pos);
}
